// SAGEEncoder_30562987278568
// MI455X (gfx1250) — matrix operand order repaired and checked by reading, not compiled, not run on hardware
//
#include <hip/hip_runtime.h>
#include <math.h>

typedef __attribute__((ext_vector_type(16))) __bf16 v16bf;
typedef __attribute__((ext_vector_type(8)))  __bf16 v8bf;
typedef __attribute__((ext_vector_type(8)))  float  v8f;

#define N_NODES 100000
#define K_DIM   128   // input dim of both layers

// ---------------------------------------------------------------------------
// Weight pre-shuffle: W [128][DOUT] f32 (row-major)  ->  bf16 fragment-major
// F[w][kb][lane][j], so each lane's B operand for (n-tile w, k-block kb) is a
// single contiguous 32B chunk (one v16bf load).
// B 32x16 bf16 layout (ISA 7.12.2): lane n=lane%16, h=lane/16;
//   element j holds K = kb*32 + h*8 + j (j<8), kb*32 + 16 + h*8 + (j-8) (j>=8).
// ---------------------------------------------------------------------------
__global__ void shuffle_weights_kernel(const float* __restrict__ W,
                                       __bf16* __restrict__ F, int DOUT) {
  int f = blockIdx.x * blockDim.x + threadIdx.x;
  if (f >= 128 * DOUT) return;
  int j    = f & 15;
  int lane = (f >> 4) & 31;
  int kb   = (f >> 9) & 3;
  int w    = f >> 11;
  int n    = w * 16 + (lane & 15);
  int h    = lane >> 4;
  int k    = kb * 32 + ((j & 8) << 1) + h * 8 + (j & 7);
  F[f] = (__bf16)W[k * DOUT + n];
}

__global__ void fill_neginf_kernel(float* __restrict__ p, unsigned int n) {
  unsigned int i = blockIdx.x * blockDim.x + threadIdx.x;
  if (i < n) p[i] = -__builtin_inff();
}

// ---------------------------------------------------------------------------
// Edge scatter: agg[dst] = max(agg[dst], x[src]) per feature.
// One thread = one (edge, float4 chunk): 1 coalesced 16B gather (x/h are
// L2-resident: 51MB < 192MB L2) + 4 native fire-and-forget
// GLOBAL_ATOMIC_MAX_NUM_F32 (STOREcnt, no return traffic).
// ---------------------------------------------------------------------------
template<int LOGC>
__global__ __launch_bounds__(256)
void scatter_max_kernel(const float* __restrict__ x,
                        const int* __restrict__ src,
                        const int* __restrict__ dst,
                        float* __restrict__ agg,
                        long long nWork) {
  long long gid = (long long)blockIdx.x * blockDim.x + threadIdx.x;
  if (gid >= nWork) return;
  const int D = 4 << LOGC;
  long long e = gid >> LOGC;
  int c = (int)(gid & ((1 << LOGC) - 1));
  int s = src[e];
  int d = dst[e];
  const float4 v = *(const float4*)(x + (size_t)s * D + c * 4);
  float* p = agg + (size_t)d * D + c * 4;
  asm volatile(
      "global_atomic_max_num_f32 %0, %1, off\n\t"
      "global_atomic_max_num_f32 %0, %2, off offset:4\n\t"
      "global_atomic_max_num_f32 %0, %3, off offset:8\n\t"
      "global_atomic_max_num_f32 %0, %4, off offset:12"
      :
      : "v"(p), "v"(v.x), "v"(v.y), "v"(v.z), "v"(v.w)
      : "memory");
}

// ---------------------------------------------------------------------------
// Fused SAGE layer: out = maybe_relu(L2norm(fix(agg)@Wl + x@Wr + b)).
// Block = 16 nodes x DOUT cols, one wave per 16-col tile.
// A rows (shared by ALL waves) are staged once through LDS as bf16 with the
// -inf fixup applied once; each lane then builds its A fragment from two
// ds_load_b128 (the A layout is two contiguous 8-elem runs per k-block).
// B fragments are single contiguous 32B global loads (pre-shuffled, L0-hot).
// K=128 -> 4 k-blocks x two v_wmma_f32_16x16x32_bf16 sharing one accumulator.
// Row sum-of-squares reduced through LDS in fixed order -> deterministic.
// ---------------------------------------------------------------------------
template<int DOUT, bool RELU>
__global__ __launch_bounds__(DOUT * 2)
void sage_wmma_kernel(const float* __restrict__ agg,   // [N,128], -inf = no edge
                      const float* __restrict__ xr,    // [N,128] root features
                      const __bf16* __restrict__ WlF,  // frag-major [DOUT/16][4][32][16]
                      const __bf16* __restrict__ WrF,
                      const float* __restrict__ bias,  // [DOUT]
                      float* __restrict__ out) {       // [N,DOUT]
  constexpr int W = DOUT / 16;        // waves per block
  constexpr int STRIDE = 136;         // bf16/row: 128 + 8 pad (272B -> bank-skew 4)
  __shared__ __bf16 aggS[16 * STRIDE];
  __shared__ __bf16 xS[16 * STRIDE];
  __shared__ float prt[W][2][8][16];  // [wave][h][r][n] partial squares
  __shared__ float nrm[16];

  const int tid  = threadIdx.x;
  const int lane = tid & 31;
  const int wave = tid >> 5;
  const int n    = lane & 15;
  const int h    = lane >> 4;
  const int rowBase = blockIdx.x * 16;

  // --- Stage 16 rows x 128 f32 of agg and x -> bf16 in LDS (coalesced) ---
  // 512 float4 chunks per matrix; blockDim (256 or 128) divides 512 evenly.
  for (int c = tid; c < 512; c += DOUT * 2) {
    const int m = c >> 5;   // row 0..15
    const int q = c & 31;   // float4 chunk 0..31
    float4 fa = *(const float4*)(agg + (size_t)(rowBase + m) * K_DIM + q * 4);
    // PyG scatter-max semantics: nodes with no in-edges aggregate to 0
    fa.x = (fa.x < -1e37f) ? 0.f : fa.x;
    fa.y = (fa.y < -1e37f) ? 0.f : fa.y;
    fa.z = (fa.z < -1e37f) ? 0.f : fa.z;
    fa.w = (fa.w < -1e37f) ? 0.f : fa.w;
    __bf16* da = &aggS[m * STRIDE + q * 4];
    da[0] = (__bf16)fa.x; da[1] = (__bf16)fa.y;
    da[2] = (__bf16)fa.z; da[3] = (__bf16)fa.w;
    const float4 fx = *(const float4*)(xr + (size_t)(rowBase + m) * K_DIM + q * 4);
    __bf16* dx = &xS[m * STRIDE + q * 4];
    dx[0] = (__bf16)fx.x; dx[1] = (__bf16)fx.y;
    dx[2] = (__bf16)fx.z; dx[3] = (__bf16)fx.w;
  }
  __syncthreads();

  const __bf16* aRow = &aggS[n * STRIDE];  // A-row M = lane%16
  const __bf16* xRow = &xS[n * STRIDE];
  const __bf16* wl = WlF + ((size_t)(wave * 4) * 32 + lane) * 16;
  const __bf16* wr = WrF + ((size_t)(wave * 4) * 32 + lane) * 16;

  v8f c = {};
#pragma unroll
  for (int kb = 0; kb < 4; ++kb) {
    // A fragment: lo half = k in [kb*32+h*8, +8), hi half = +16 (ISA 7.12.2)
    v8bf aAlo = *(const v8bf*)(aRow + kb * 32 + h * 8);
    v8bf aAhi = *(const v8bf*)(aRow + kb * 32 + 16 + h * 8);
    v16bf aA = __builtin_shufflevector(aAlo, aAhi, 0, 1, 2, 3, 4, 5, 6, 7,
                                       8, 9, 10, 11, 12, 13, 14, 15);
    v8bf aXlo = *(const v8bf*)(xRow + kb * 32 + h * 8);
    v8bf aXhi = *(const v8bf*)(xRow + kb * 32 + 16 + h * 8);
    v16bf aX = __builtin_shufflevector(aXlo, aXhi, 0, 1, 2, 3, 4, 5, 6, 7,
                                       8, 9, 10, 11, 12, 13, 14, 15);
    v16bf bL = *(const v16bf*)(wl + (size_t)kb * 32 * 16);
    v16bf bR = *(const v16bf*)(wr + (size_t)kb * 32 * 16);
    c = __builtin_amdgcn_wmma_f32_16x16x32_bf16(false, aA, false, bL,
                                                (short)0, c, false, false);
    c = __builtin_amdgcn_wmma_f32_16x16x32_bf16(false, aX, false, bR,
                                                (short)0, c, false, false);
  }

  const float bv = bias[wave * 16 + n];
  float vals[8];
#pragma unroll
  for (int r = 0; r < 8; ++r) {
    float t = c[r] + bv;
    vals[r] = t;
    prt[wave][h][r][n] = t * t;  // deterministic: no float atomics
  }
  __syncthreads();

  if (tid < 16) {
    const int hh = tid >> 3, rr = tid & 7;
    float s = 0.f;
#pragma unroll
    for (int w = 0; w < W; ++w)
      for (int nn = 0; nn < 16; ++nn)
        s += prt[w][hh][rr][nn];
    nrm[tid] = fmaxf(sqrtf(s), 1e-12f);  // F.normalize eps clamp
  }
  __syncthreads();

#pragma unroll
  for (int r = 0; r < 8; ++r) {
    float o = vals[r] / nrm[h * 8 + r];
    if (RELU) o = fmaxf(o, 0.f);
    out[(size_t)(rowBase + h * 8 + r) * DOUT + wave * 16 + n] = o;
  }
}

// ---------------------------------------------------------------------------
// Launch
// ---------------------------------------------------------------------------
extern "C" void kernel_launch(void* const* d_in, const int* in_sizes, int n_in,
                              void* d_out, int out_size, void* d_ws, size_t ws_size,
                              hipStream_t stream) {
  const float* x   = (const float*)d_in[0];
  const int*   ei  = (const int*)d_in[1];
  const float* Wl1 = (const float*)d_in[2];
  const float* Wr1 = (const float*)d_in[3];
  const float* b1  = (const float*)d_in[4];
  const float* Wl2 = (const float*)d_in[5];
  const float* Wr2 = (const float*)d_in[6];
  const float* b2  = (const float*)d_in[7];
  float* out = (float*)d_out;

  const long long E = in_sizes[1] / 2;
  const int* src = ei;
  const int* dst = ei + E;

  // Workspace: agg (N*128 f32, reused as N*64 for layer 2), h (N*128 f32),
  // then bf16 fragment-major weight copies. ~102.5 MB total.
  char* ws = (char*)d_ws;
  float*  agg  = (float*)ws;
  float*  hbuf = (float*)(ws + (size_t)N_NODES * 128 * 4);
  __bf16* wbf  = (__bf16*)(ws + 2ull * N_NODES * 128 * 4);
  __bf16* Wl1F = wbf;
  __bf16* Wr1F = wbf + 128 * 128;
  __bf16* Wl2F = wbf + 2 * 128 * 128;
  __bf16* Wr2F = wbf + 2 * 128 * 128 + 128 * 64;

  shuffle_weights_kernel<<<(128 * 128 + 255) / 256, 256, 0, stream>>>(Wl1, Wl1F, 128);
  shuffle_weights_kernel<<<(128 * 128 + 255) / 256, 256, 0, stream>>>(Wr1, Wr1F, 128);
  shuffle_weights_kernel<<<(128 * 64 + 255) / 256, 256, 0, stream>>>(Wl2, Wl2F, 64);
  shuffle_weights_kernel<<<(128 * 64 + 255) / 256, 256, 0, stream>>>(Wr2, Wr2F, 64);

  // ---- Layer 1: d=128 ----
  {
    unsigned int nElem = (unsigned int)N_NODES * 128u;
    fill_neginf_kernel<<<(nElem + 255) / 256, 256, 0, stream>>>(agg, nElem);
    long long work = E * 32;  // 32 float4 chunks per edge
    scatter_max_kernel<5><<<(unsigned int)((work + 255) / 256), 256, 0, stream>>>(
        x, src, dst, agg, work);
    sage_wmma_kernel<128, true><<<N_NODES / 16, 256, 0, stream>>>(
        agg, x, Wl1F, Wr1F, b1, hbuf);
  }

  // ---- Layer 2: d_out=64 ----
  {
    unsigned int nElem = (unsigned int)N_NODES * 64u;
    fill_neginf_kernel<<<(nElem + 255) / 256, 256, 0, stream>>>(agg, nElem);
    long long work = E * 16;  // 16 float4 chunks per edge
    scatter_max_kernel<4><<<(unsigned int)((work + 255) / 256), 256, 0, stream>>>(
        hbuf, src, dst, agg, work);
    sage_wmma_kernel<64, false><<<N_NODES / 16, 128, 0, stream>>>(
        agg, hbuf, Wl2F, Wr2F, b2, out);
  }
}
